// DeepSeekGate_7267084664931
// MI455X (gfx1250) — matrix operand order repaired and checked by reading, not compiled, not run on hardware
//
#include <hip/hip_runtime.h>
#include <hip/hip_bf16.h>
#include <math.h>

// Problem constants (reference: T=16384, D=7168, N=256)
#define DIM   7168
#define NEXP  256
#define NGRP  8          // 8 groups of 32 experts
#define TOPG  4
#define NACT  8
#define RSCALE 2.5f

typedef __attribute__((ext_vector_type(16))) _Float16 v16h;
typedef __attribute__((ext_vector_type(8)))  _Float16 v8h;
typedef __attribute__((ext_vector_type(4)))  _Float16 v4h;
typedef __attribute__((ext_vector_type(8)))  float    v8f;
typedef __attribute__((ext_vector_type(4)))  float    v4f;

// ---------------------------------------------------------------------------
// Kernel 0: fp32 -> f16 weight conversion into workspace (3.67 MB, L2-resident)
// ---------------------------------------------------------------------------
__global__ void cvt_w_kernel(const float* __restrict__ w,
                             _Float16* __restrict__ w16, int n)
{
    int i = (blockIdx.x * blockDim.x + threadIdx.x) * 4;
    if (i >= n) return;
    v4f f = *(const v4f*)(w + i);
    v4h h;
    h.x = (_Float16)f.x; h.y = (_Float16)f.y;
    h.z = (_Float16)f.z; h.w = (_Float16)f.w;
    *(v4h*)(w16 + i) = h;
}

// ---------------------------------------------------------------------------
// Kernel 1: logits = x * W^T via v_wmma_f32_16x16x32_f16, sigmoid -> scores
// Block: 256 threads = 8 waves; tile M=64 rows x N=256 experts.
// Wave (wm,wn): wm=wave/4 in {0,1} -> rows [wm*32, wm*32+32)
//               wn=wave%4 in {0..3} -> cols [wn*64, wn*64+64)
// Each wave: 2 M-tiles x 4 N-tiles of 16x16 f32 accumulators.
//
// Fragment layouts used here (same contiguous K order for A and B):
//   A (16x32 f16): lane (h=lane/16, m=lane%16) holds K in
//       h==0 -> {0..15}, h==1 -> {16..31}  (two 16B chunks)
//   B (32x16 f16): lane (h, n=lane%16) holds K = h*16 .. h*16+15 (contiguous 32B)
//   C (16x16 f32): VGPR r -> M = r + (lane>=16 ? 8 : 0), N = lane%16
// ---------------------------------------------------------------------------
#define AROW 40   // f16 elems per LDS A row: 32 data + 8 pad = 80 B (16B-aligned, bank-friendly)

__global__ __launch_bounds__(256)
void gate_gemm_kernel(const float* __restrict__ x,
                      const _Float16* __restrict__ w16,
                      float* __restrict__ scores)
{
    __shared__ __align__(16) _Float16 lA[64 * AROW];   // 5 KB

    const int tid  = threadIdx.x;
    const int lane = tid & 31;
    const int wave = tid >> 5;
    const int wm   = wave >> 2;          // 0..1
    const int wn   = wave & 3;           // 0..3
    const int row0 = blockIdx.x * 64;

    const int h  = lane >> 4;            // half-wave (K-split of A/B layout)
    const int ln = lane & 15;            // M (for A) / N (for B) index

    v8f acc[2][4];
    #pragma unroll
    for (int i = 0; i < 2; ++i)
        #pragma unroll
        for (int j = 0; j < 4; ++j)
            acc[i][j] = (v8f){0.f,0.f,0.f,0.f,0.f,0.f,0.f,0.f};

    // A staging: 64 rows x 32 k fp32 -> f16 LDS; each thread converts 8 elems
    const int srow = tid >> 2;           // 0..63
    const int sseg = (tid & 3) * 8;      // 0,8,16,24
    const float*  xsrc = x + (size_t)(row0 + srow) * DIM + sseg;
    _Float16*     ldst = lA + srow * AROW + sseg;

    // B bases: per N-tile nt, lane covers expert col = wn*64 + nt*16 + ln,
    // contiguous K run starting at h*16.
    const _Float16* bbase[4];
    #pragma unroll
    for (int nt = 0; nt < 4; ++nt)
        bbase[nt] = w16 + (size_t)(wn * 64 + nt * 16 + ln) * DIM + h * 16;

    for (int k0 = 0; k0 < DIM; k0 += 32) {
        // ---- stage x chunk into LDS as f16 ----
        v4f f0 = *(const v4f*)(xsrc + k0);
        v4f f1 = *(const v4f*)(xsrc + k0 + 4);
        v8h hs;
        hs[0] = (_Float16)f0.x; hs[1] = (_Float16)f0.y;
        hs[2] = (_Float16)f0.z; hs[3] = (_Float16)f0.w;
        hs[4] = (_Float16)f1.x; hs[5] = (_Float16)f1.y;
        hs[6] = (_Float16)f1.z; hs[7] = (_Float16)f1.w;
        *(v8h*)ldst = hs;
        __syncthreads();

        // ---- A fragments from LDS (contiguous K = h*16..h*16+15, two 16B chunks) ----
        union { v16h v; v8h p[2]; } afr[2];
        #pragma unroll
        for (int mt = 0; mt < 2; ++mt) {
            const _Float16* ap = lA + (wm * 32 + mt * 16 + ln) * AROW + h * 16;
            afr[mt].p[0] = *(const v8h*)(ap);        // k: h? 16..23 : 0..7
            afr[mt].p[1] = *(const v8h*)(ap + 8);    // k: h? 24..31 : 8..15
        }

        // ---- B fragments direct from L2-resident f16 weights (contiguous 32B) ----
        v16h bfr[4];
        #pragma unroll
        for (int nt = 0; nt < 4; ++nt)
            bfr[nt] = *(const v16h*)(bbase[nt] + k0);  // K = k0 + h*16 .. +15

        // ---- 8 WMMAs: 16x16x32 f16 -> f32 ----
        #pragma unroll
        for (int mt = 0; mt < 2; ++mt)
            #pragma unroll
            for (int nt = 0; nt < 4; ++nt)
                acc[mt][nt] = __builtin_amdgcn_wmma_f32_16x16x32_f16(
                    false, afr[mt].v, false, bfr[nt],
                    (short)0, acc[mt][nt], false, false);

        __syncthreads();
    }

    // ---- epilogue: sigmoid, write scores[T, 256] ----
    #pragma unroll
    for (int mt = 0; mt < 2; ++mt) {
        #pragma unroll
        for (int nt = 0; nt < 4; ++nt) {
            const int col = wn * 64 + nt * 16 + ln;
            #pragma unroll
            for (int r = 0; r < 8; ++r) {
                const int grow = row0 + wm * 32 + mt * 16 + r + (h ? 8 : 0);
                const float z = acc[mt][nt][r];
                scores[(size_t)grow * NEXP + col] = 1.f / (1.f + __expf(-z));
            }
        }
    }
}

// ---------------------------------------------------------------------------
// Kernel 2: grouped top-k routing. One wave32 per token row, all in registers.
// Lane owns experts [lane*8, lane*8+8); group g = lane/4 (32 experts/group).
// ---------------------------------------------------------------------------
__global__ __launch_bounds__(256)
void gate_route_kernel(const float* __restrict__ scores,
                       const float* __restrict__ bias,
                       float* __restrict__ wout,
                       int* __restrict__ iout, int T)
{
    const int lane = threadIdx.x & 31;
    const int row  = blockIdx.x * (blockDim.x >> 5) + (threadIdx.x >> 5);
    if (row >= T) return;

    const float* sp = scores + (size_t)row * NEXP + lane * 8;
    const float* bp = bias + lane * 8;
    v4f a0 = *(const v4f*)(sp);
    v4f a1 = *(const v4f*)(sp + 4);
    v4f b0 = *(const v4f*)(bp);
    v4f b1 = *(const v4f*)(bp + 4);

    float s[8], r[8];
    #pragma unroll
    for (int j = 0; j < 4; ++j) {
        s[j]     = a0[j]; r[j]     = a0[j] + b0[j];
        s[4 + j] = a1[j]; r[4 + j] = a1[j] + b1[j];
    }

    // per-lane top-2 of routing values
    float m1 = -INFINITY, m2 = -INFINITY;
    #pragma unroll
    for (int j = 0; j < 8; ++j) {
        float v = r[j];
        if (v > m1) { m2 = m1; m1 = v; }
        else if (v > m2) { m2 = v; }
    }
    // merge top-2 across the 4 lanes of each group (xor 1, 2)
    #pragma unroll
    for (int mask = 1; mask <= 2; mask <<= 1) {
        float o1 = __shfl_xor(m1, mask, 32);
        float o2 = __shfl_xor(m2, mask, 32);
        float n1 = fmaxf(m1, o1);
        float n2 = fmaxf(fminf(m1, o1), fmaxf(m2, o2));
        m1 = n1; m2 = n2;
    }
    const float gs = m1 + m2;          // group score (same in all 4 lanes)
    const int   g  = lane >> 2;

    // broadcast all 8 group scores, rank my group (stable lowest-index ties)
    float gsg[8];
    #pragma unroll
    for (int j = 0; j < 8; ++j) gsg[j] = __shfl(gs, j * 4, 32);
    int rank = 0;
    #pragma unroll
    for (int j = 0; j < 8; ++j)
        if (j != g && (gsg[j] > gs || (gsg[j] == gs && j < g))) rank++;
    if (rank >= TOPG) {
        #pragma unroll
        for (int j = 0; j < 8; ++j) r[j] = -INFINITY;
    }

    // iterative top-8 via wave argmax reduction (carry value, index, raw score)
    float wv[NACT]; int wi[NACT]; float wsum = 0.f;
    #pragma unroll
    for (int it = 0; it < NACT; ++it) {
        float lv = -INFINITY, lsc = 0.f; int lj = 0;
        #pragma unroll
        for (int j = 0; j < 8; ++j)
            if (r[j] > lv) { lv = r[j]; lj = j; lsc = s[j]; }
        float bv = lv, bs = lsc; int bi = lane * 8 + lj;
        #pragma unroll
        for (int mask = 16; mask >= 1; mask >>= 1) {
            float ov = __shfl_xor(bv, mask, 32);
            int   oi = __shfl_xor(bi, mask, 32);
            float os = __shfl_xor(bs, mask, 32);
            if (ov > bv || (ov == bv && oi < bi)) { bv = ov; bi = oi; bs = os; }
        }
        wv[it] = bs; wi[it] = bi; wsum += bs;
        if ((bi >> 3) == lane) {               // owner retires the element
            const int k = bi & 7;
            #pragma unroll
            for (int j = 0; j < 8; ++j)
                if (j == k) r[j] = -INFINITY;
        }
    }

    const float inv = RSCALE / fmaxf(wsum, 1e-9f);
    if (lane == 0) {
        #pragma unroll
        for (int it = 0; it < NACT; ++it) {
            wout[(size_t)row * NACT + it] = wv[it] * inv;
            iout[(size_t)row * NACT + it] = wi[it];
        }
    }
}

// ---------------------------------------------------------------------------
extern "C" void kernel_launch(void* const* d_in, const int* in_sizes, int n_in,
                              void* d_out, int out_size, void* d_ws, size_t ws_size,
                              hipStream_t stream)
{
    const float* x    = (const float*)d_in[0];   // [T, 7168] fp32
    const float* w    = (const float*)d_in[1];   // [256, 7168] fp32
    const float* bias = (const float*)d_in[2];   // [256] fp32
    const int T = in_sizes[0] / DIM;             // 16384

    float* out      = (float*)d_out;
    float* wout     = out;                                   // [T,8] f32
    int*   iout     = (int*)(out + (size_t)T * NACT);        // [T,8] i32 (raw bits)
    float* scores   = out + (size_t)T * NACT * 2;            // [T,256] f32
    _Float16* w16   = (_Float16*)d_ws;                       // 256*7168 f16 = 3.67 MB

    const int nw = in_sizes[1];                  // 256*7168
    cvt_w_kernel<<<(nw / 4 + 255) / 256, 256, 0, stream>>>(w, w16, nw);
    gate_gemm_kernel<<<T / 64, 256, 0, stream>>>(x, w16, scores);
    gate_route_kernel<<<(T + 7) / 8, 256, 0, stream>>>(scores, bias, wout, iout, T);
}
